// TransformerEncoderLayer_89163521065783
// MI455X (gfx1250) — matrix operand order repaired and checked by reading, not compiled, not run on hardware
//
#include <hip/hip_runtime.h>
#include <hip/hip_bf16.h>
#include <math.h>

// ---------------------------------------------------------------------------
// Types for CDNA5 WMMA (wave32): V_WMMA_F32_16X16X32_BF16
// ---------------------------------------------------------------------------
typedef __attribute__((ext_vector_type(16))) __bf16 v16bf;
typedef __attribute__((ext_vector_type(8)))  __bf16 v8bf;
typedef __attribute__((ext_vector_type(8)))  float  v8f;

#define WMMA_BF16(A, B, C) \
  __builtin_amdgcn_wmma_f32_16x16x32_bf16(false, (A), false, (B), (short)0, (C), false, false)

// Async global->LDS (CDNA5 GLOBAL_LOAD_ASYNC_TO_LDS_B128, ASYNCcnt-tracked).
#if __has_builtin(__builtin_amdgcn_global_load_async_to_lds_b128) && \
    __has_builtin(__builtin_amdgcn_s_wait_asynccnt)
#define HAVE_ASYNC_LDS 1
#else
#define HAVE_ASYNC_LDS 0
#endif

#if HAVE_ASYNC_LDS
// Builtin signature (from hipcc diagnostic): params are pointers to
// 'int __attribute__((vector_size(16)))' in AS1 (global) / AS3 (LDS).
typedef int v4i_g __attribute__((vector_size(16)));
typedef __attribute__((address_space(1))) v4i_g* gptr_b128;
typedef __attribute__((address_space(3))) v4i_g* sptr_b128;

static __device__ __forceinline__ void async_b128(const void* g, void* l) {
  // AS1 via integer round-trip (flat==global addr); AS3 via low-32 truncation
  // (ISA 10.2: LDS flat address truncates to LDS byte address).
  __builtin_amdgcn_global_load_async_to_lds_b128(
      (gptr_b128)(unsigned long long)g,
      (sptr_b128)(unsigned int)(unsigned long long)l,
      0, 0);
}
#endif

static __device__ __forceinline__ v16bf pack2(v8bf lo, v8bf hi) {
  v16bf r;
#pragma unroll
  for (int i = 0; i < 8; ++i) { r[i] = lo[i]; r[i + 8] = hi[i]; }
  return r;
}

static __device__ __forceinline__ v8f zero8() {
  v8f z;
#pragma unroll
  for (int i = 0; i < 8; ++i) z[i] = 0.0f;
  return z;
}

// DPP butterfly max over a 16-lane row (pure VALU, no LDS traffic).
template <int CTRL>
static __device__ __forceinline__ float dpp_max_step(float v) {
  int i = __builtin_bit_cast(int, v);
  int p = __builtin_amdgcn_update_dpp(i, i, CTRL, 0xF, 0xF, true);
  return fmaxf(v, __builtin_bit_cast(float, p));
}
static __device__ __forceinline__ float rowmax16(float v) {
  v = dpp_max_step<0xB1>(v);   // quad_perm(1,0,3,2)  : xor 1
  v = dpp_max_step<0x4E>(v);   // quad_perm(2,3,0,1)  : xor 2
  v = dpp_max_step<0x141>(v);  // row_half_mirror     : combine 4-groups
  v = dpp_max_step<0x140>(v);  // row_mirror          : combine 8-groups
  return v;
}

// ---------------------------------------------------------------------------
// Problem constants (match reference)
// ---------------------------------------------------------------------------
constexpr int CB  = 8;
constexpr int CS  = 1024;
constexpr int CD  = 1024;
constexpr int CH  = 16;
constexpr int CHD = 64;
constexpr int CFF = 4096;
constexpr int CM  = CB * CS;   // 8192 token rows
constexpr int CQ3 = 3 * CD;    // 3072 fused qkv columns

// ---------------------------------------------------------------------------
// LayerNorm (f32 in) -> bf16 out. One block per row, 256 threads, D=1024.
// ---------------------------------------------------------------------------
__global__ __launch_bounds__(256) void layernorm_bf16_kernel(
    const float* __restrict__ x, const float* __restrict__ g,
    const float* __restrict__ bta, __bf16* __restrict__ out) {
  __shared__ float s1[256];
  __shared__ float s2[256];
  const int row = blockIdx.x;
  const int tid = threadIdx.x;
  const float* xr = x + (size_t)row * CD;
  float xv[4];
  float sum = 0.0f, sq = 0.0f;
#pragma unroll
  for (int i = 0; i < 4; ++i) {
    xv[i] = xr[tid + i * 256];
    sum += xv[i];
    sq  += xv[i] * xv[i];
  }
  s1[tid] = sum; s2[tid] = sq;
  __syncthreads();
  for (int off = 128; off > 0; off >>= 1) {
    if (tid < off) { s1[tid] += s1[tid + off]; s2[tid] += s2[tid + off]; }
    __syncthreads();
  }
  const float mu  = s1[0] * (1.0f / CD);
  const float var = s2[0] * (1.0f / CD) - mu * mu;
  const float rsd = rsqrtf(var + 1e-5f);
#pragma unroll
  for (int i = 0; i < 4; ++i) {
    const int c = tid + i * 256;
    out[(size_t)row * CD + c] = (__bf16)((xv[i] - mu) * rsd * g[c] + bta[c]);
  }
}

// ---------------------------------------------------------------------------
// Weight prep: f32 [K,N] -> bf16 transposed [N,K], so GEMM B-fragments are
// contiguous K-runs (matches the 16-bit WMMA B layout) with no LDS scatter.
// ---------------------------------------------------------------------------
__global__ void cvt_transpose_kernel(const float* __restrict__ in,
                                     __bf16* __restrict__ out, int K, int N) {
  const int total = K * N;
  for (int i = blockIdx.x * blockDim.x + threadIdx.x; i < total;
       i += gridDim.x * blockDim.x) {
    const int n = i / K;
    const int k = i - n * K;
    out[i] = (__bf16)in[(size_t)k * N + n];
  }
}

// Fused QKV weight, transposed: out[n*D + d] = W_t[h, d, kk]
// (q part pre-scaled by 1/sqrt(HD) so attention scores need no scaling).
__global__ void build_wqkvT_kernel(const float* __restrict__ Wq,
                                   const float* __restrict__ Wk,
                                   const float* __restrict__ Wv,
                                   __bf16* __restrict__ out) {
  const int total = CQ3 * CD;
  for (int i = blockIdx.x * blockDim.x + threadIdx.x; i < total;
       i += gridDim.x * blockDim.x) {
    const int n  = i >> 10;            // 0..3071
    const int d  = i & (CD - 1);
    const int t  = n >> 10;            // 0:q 1:k 2:v
    const int hh = (n >> 6) & (CH - 1);
    const int kk = n & (CHD - 1);
    const float* W = (t == 0) ? Wq : ((t == 1) ? Wk : Wv);
    float v = W[((size_t)hh * CD + d) * CHD + kk];
    if (t == 0) v *= 0.125f;           // 1/sqrt(64)
    out[i] = (__bf16)v;
  }
}

__global__ void build_bqkv_kernel(const float* __restrict__ bq,
                                  const float* __restrict__ bk,
                                  const float* __restrict__ bv,
                                  float* __restrict__ out) {
  const int i = blockIdx.x * blockDim.x + threadIdx.x;
  if (i < CQ3) {
    const int t = i >> 10;
    const int r = i & 1023;
    out[i] = (t == 0) ? (bq[r] * 0.125f) : ((t == 1) ? bk[r] : bv[r]);
  }
}

// ---------------------------------------------------------------------------
// Tiled bf16 WMMA GEMM with TRANSPOSED weights:
//   out[M,N] = epi( A[M,K] * Bt[N,K]^T + bias[N] )
// Block tile 128x128, K-step 32. 8 waves (2 in M x 4 in N), each wave a
// 64x32 macro-tile = 4x2 v_wmma_f32_16x16x32_bf16 tiles (8 WMMA / K-step).
// Async path: double-buffered LDS, GLOBAL_LOAD_ASYNC_TO_LDS_B128 prefetches
// the next K-tile while the current tile's WMMAs run; each wave waits only
// asynccnt<=4 (its own current-tile copies, in-order) before the barrier.
// EPI: 0 = bias, 1 = bias+GELU(exact), 2 = bias+residual,
//      3 = QKV (q,k -> out[M,3072]; v -> vt[b,h,hd,s] transposed)
// ---------------------------------------------------------------------------
template <int EPI, typename OutT>
__global__ __launch_bounds__(256) void gemm_bt_kernel(
    const __bf16* __restrict__ A, const __bf16* __restrict__ Bt,
    const float* __restrict__ bias, const float* __restrict__ res,
    OutT* __restrict__ out, __bf16* __restrict__ vt, int M, int N, int K) {
#if HAVE_ASYNC_LDS
  __shared__ __align__(64) __bf16 As[2][128 * 32];   // [m][k], double buffered
  __shared__ __align__(64) __bf16 Bs[2][128 * 32];   // [n][k], double buffered
#else
  __shared__ __align__(64) __bf16 As[1][128 * 32];
  __shared__ __align__(64) __bf16 Bs[1][128 * 32];
#endif

  const int tid  = threadIdx.x;
  const int w    = tid >> 5;
  const int lane = tid & 31;
  const int lr   = lane & 15;
  const int lg   = lane >> 4;
  const int wm   = w >> 2;     // 0..1 -> M offset wm*64
  const int wn   = w & 3;      // 0..3 -> N offset wn*32
  const int rowBase = blockIdx.y * 128;
  const int colBase = blockIdx.x * 128;

  v8f acc[4][2];
#pragma unroll
  for (int i = 0; i < 4; ++i)
#pragma unroll
    for (int j = 0; j < 2; ++j) acc[i][j] = zero8();

  const int srow  = tid >> 1;          // 0..127
  const int spart = (tid & 1) * 16;    // 0 or 16
  const int soff  = srow * 32 + spart; // LDS element offset for staging

#if HAVE_ASYNC_LDS
  const int nk = K >> 5;
  // Prologue: issue async copies for tile 0.
  {
    const __bf16* ag = A  + (size_t)(rowBase + srow) * K + spart;
    const __bf16* bg = Bt + (size_t)(colBase + srow) * K + spart;
    async_b128(ag,     &As[0][soff]);
    async_b128(ag + 8, &As[0][soff + 8]);
    async_b128(bg,     &Bs[0][soff]);
    async_b128(bg + 8, &Bs[0][soff + 8]);
  }
  for (int kt = 0; kt < nk; ++kt) {
    const int cur = kt & 1;
    if (kt + 1 < nk) {
      const int k1 = (kt + 1) * 32;
      const __bf16* ag = A  + (size_t)(rowBase + srow) * K + k1 + spart;
      const __bf16* bg = Bt + (size_t)(colBase + srow) * K + k1 + spart;
      async_b128(ag,     &As[cur ^ 1][soff]);
      async_b128(ag + 8, &As[cur ^ 1][soff + 8]);
      async_b128(bg,     &Bs[cur ^ 1][soff]);
      async_b128(bg + 8, &Bs[cur ^ 1][soff + 8]);
      __builtin_amdgcn_s_wait_asynccnt(4);  // current tile done, next in flight
    } else {
      __builtin_amdgcn_s_wait_asynccnt(0);
    }
    __syncthreads();
    const __bf16* asb = As[cur];
    const __bf16* bsb = Bs[cur];
#else
  for (int k0 = 0; k0 < K; k0 += 32) {
    const __bf16* ag = A  + (size_t)(rowBase + srow) * K + k0 + spart;
    const __bf16* bg = Bt + (size_t)(colBase + srow) * K + k0 + spart;
    *(v8bf*)&As[0][soff]     = *(const v8bf*)ag;
    *(v8bf*)&As[0][soff + 8] = *(const v8bf*)(ag + 8);
    *(v8bf*)&Bs[0][soff]     = *(const v8bf*)bg;
    *(v8bf*)&Bs[0][soff + 8] = *(const v8bf*)(bg + 8);
    if (k0 + 32 < K) {
      __builtin_prefetch(ag + 32, 0, 1);
      __builtin_prefetch(bg + 32, 0, 1);
    }
    __syncthreads();
    const __bf16* asb = As[0];
    const __bf16* bsb = Bs[0];
#endif

    v16bf af[4], bfr[2];
#pragma unroll
    for (int i = 0; i < 4; ++i) {
      const int r = wm * 64 + i * 16 + lr;
      af[i] = *(const v16bf*)&asb[r * 32 + lg * 16];
    }
#pragma unroll
    for (int j = 0; j < 2; ++j) {
      const int c = wn * 32 + j * 16 + lr;
      bfr[j] = *(const v16bf*)&bsb[c * 32 + lg * 16];
    }
#pragma unroll
    for (int i = 0; i < 4; ++i)
#pragma unroll
      for (int j = 0; j < 2; ++j)
        acc[i][j] = WMMA_BF16(af[i], bfr[j], acc[i][j]);
    __syncthreads();
  }

  // Epilogue. C layout: VGPR e -> row e + 8*lg ; col = lr within 16x16 tile.
#pragma unroll
  for (int i = 0; i < 4; ++i) {
#pragma unroll
    for (int j = 0; j < 2; ++j) {
#pragma unroll
      for (int e = 0; e < 8; ++e) {
        const int grow = rowBase + wm * 64 + i * 16 + lg * 8 + e;
        const int gcol = colBase + wn * 32 + j * 16 + lr;
        float v = acc[i][j][e] + bias[gcol];
        if constexpr (EPI == 1)
          v = 0.5f * v * (1.0f + erff(v * 0.70710678118654752f));
        if constexpr (EPI == 2)
          v += res[(size_t)grow * N + gcol];
        if constexpr (EPI == 3) {
          if (gcol < 2 * CD) {
            out[(size_t)grow * N + gcol] = (OutT)v;   // q | k
          } else {                                     // v -> [b,h,hd,s]
            const int hh = (gcol >> 6) & (CH - 1);
            const int kk = gcol & (CHD - 1);
            const int bb = grow >> 10;
            const int ss = grow & (CS - 1);
            vt[(((size_t)bb * CH + hh) * CHD + kk) * CS + ss] = (__bf16)v;
          }
        } else {
          out[(size_t)grow * N + gcol] = (OutT)v;
        }
      }
    }
  }
}

// ---------------------------------------------------------------------------
// Flash attention. q,k from fused qkv[M,3072] bf16; v from vt[b,h,hd,s] bf16.
// Grid: (S/128, B*H). 8 waves; wave w owns 16 q-rows, streams 32-key blocks.
// Softmax denominator computed by WMMA against an appended ones-subtile of V
// ([V | 1] augmented matrix); row max via DPP butterflies (no LDS shuffles).
// K/V tiles staged via async global->LDS when available.
// Writes obuf[M, 1024] bf16 = concat-head attention output.
// ---------------------------------------------------------------------------
__global__ __launch_bounds__(256) void attn_kernel(
    const __bf16* __restrict__ qkv, const __bf16* __restrict__ vt,
    __bf16* __restrict__ obuf) {
  __shared__ __align__(64) __bf16 Kt[32 * 64];       // [key][dim]
  __shared__ __align__(64) __bf16 Vt[80 * 32];       // [dim][key]; rows 64..79 = ones
  __shared__ __align__(64) __bf16 Ps[8][16 * 32];    // per-wave P scratch

  const int tid  = threadIdx.x;
  const int w    = tid >> 5;
  const int lane = tid & 31;
  const int lr   = lane & 15;
  const int lg   = lane >> 4;

  const int bh = blockIdx.y;
  const int b  = bh >> 4;
  const int h  = bh & (CH - 1);
  const int qBase = blockIdx.x * 128;

  const __bf16* qp  = qkv + (size_t)b * CS * CQ3 + h * CHD;        // q cols
  const __bf16* kp  = qp + CD;                                      // k cols
  const __bf16* vpt = vt + (size_t)(b * CH + h) * CHD * CS;         // [hd][s]

  // Ones block for the row-sum WMMA (written once; synced inside the loop).
  Vt[64 * 32 + tid * 2]     = (__bf16)1.0f;
  Vt[64 * 32 + tid * 2 + 1] = (__bf16)1.0f;

  // Q fragments (q pre-scaled by 1/sqrt(HD) at weight prep).
  const int qrow = qBase + w * 16 + lr;
  const __bf16* qr = qp + (size_t)qrow * CQ3;
  v16bf qf[2];
#pragma unroll
  for (int kc = 0; kc < 2; ++kc) {
    const int base = kc * 32 + lg * 16;
    qf[kc] = *(const v16bf*)(qr + base);
  }

  float m8[8];
  v8f oacc[4], lsum;
#pragma unroll
  for (int e = 0; e < 8; ++e) m8[e] = -1e30f;
#pragma unroll
  for (int n = 0; n < 4; ++n) oacc[n] = zero8();
  lsum = zero8();

  const int kr  = tid >> 3;         // 0..31 key row
  const int kc8 = (tid & 7) * 8;    // dim chunk for K staging
  const int vd  = tid >> 2;         // 0..63 dim row
  const int vc8 = (tid & 3) * 8;    // key chunk for V staging

  for (int t0 = 0; t0 < CS; t0 += 32) {
    const __bf16* kg = kp + (size_t)(t0 + kr) * CQ3 + kc8;
    const __bf16* vg = vpt + (size_t)vd * CS + t0 + vc8;
#if HAVE_ASYNC_LDS
    async_b128(kg, &Kt[kr * 64 + kc8]);
    async_b128(vg, &Vt[vd * 32 + vc8]);
    __builtin_amdgcn_s_wait_asynccnt(0);
#else
    *(v8bf*)&Kt[kr * 64 + kc8] = *(const v8bf*)kg;
    *(v8bf*)&Vt[vd * 32 + vc8] = *(const v8bf*)vg;
#endif
    __syncthreads();

    // Scores s[16x32] = q(16x64) . k^T : two 16-col subtiles, 2 K-chunks each.
    v8f st[2];
#pragma unroll
    for (int n = 0; n < 2; ++n) {
      st[n] = zero8();
#pragma unroll
      for (int kc = 0; kc < 2; ++kc) {
        v16bf kf = *(const v16bf*)&Kt[(n * 16 + lr) * 64 + kc * 32 + lg * 16];
        st[n] = WMMA_BF16(qf[kc], kf, st[n]);
      }
    }

    // Online softmax: row max via DPP; denominator via the ones-subtile WMMA.
    float p[2][8];
#pragma unroll
    for (int e = 0; e < 8; ++e) {
      float rm = rowmax16(fmaxf(st[0][e], st[1][e]));
      const float mn   = fmaxf(m8[e], rm);
      const float corr = __expf(m8[e] - mn);
      m8[e] = mn;
      p[0][e] = __expf(st[0][e] - mn);
      p[1][e] = __expf(st[1][e] - mn);
      lsum[e] *= corr;
#pragma unroll
      for (int n = 0; n < 4; ++n) oacc[n][e] *= corr;
    }

    // Re-layout P (C frag -> A frag) through per-wave LDS scratch.
#pragma unroll
    for (int n = 0; n < 2; ++n)
#pragma unroll
      for (int e = 0; e < 8; ++e)
        Ps[w][(lg * 8 + e) * 32 + n * 16 + lr] = (__bf16)p[n][e];
    v16bf pf = *(const v16bf*)&Ps[w][lr * 32 + lg * 16];

    // o += P(16x32) . V(32x64); lsum += P . 1 (rows 64..79 of Vt).
#pragma unroll
    for (int n = 0; n < 4; ++n) {
      v16bf vf = *(const v16bf*)&Vt[(n * 16 + lr) * 32 + lg * 16];
      oacc[n] = WMMA_BF16(pf, vf, oacc[n]);
    }
    {
      v16bf vone = *(const v16bf*)&Vt[(64 + lr) * 32 + lg * 16];
      lsum = WMMA_BF16(pf, vone, lsum);
    }
    __syncthreads();
  }

  // Normalize and write concat-head output.
#pragma unroll
  for (int n = 0; n < 4; ++n)
#pragma unroll
    for (int e = 0; e < 8; ++e) {
      const float v = oacc[n][e] / lsum[e];
      const int row = qBase + w * 16 + lg * 8 + e;
      obuf[(size_t)(b * CS + row) * CD + h * CHD + n * 16 + lr] = (__bf16)v;
    }
}

// ---------------------------------------------------------------------------
// Orchestration
// ---------------------------------------------------------------------------
extern "C" void kernel_launch(void* const* d_in, const int* in_sizes, int n_in,
                              void* d_out, int out_size, void* d_ws,
                              size_t ws_size, hipStream_t stream) {
  (void)in_sizes; (void)n_in; (void)out_size; (void)ws_size;
  const float* x     = (const float*)d_in[0];
  const float* ln1_g = (const float*)d_in[1];
  const float* ln1_b = (const float*)d_in[2];
  const float* ln2_g = (const float*)d_in[3];
  const float* ln2_b = (const float*)d_in[4];
  const float* Wq    = (const float*)d_in[5];
  const float* bq    = (const float*)d_in[6];
  const float* Wk    = (const float*)d_in[7];
  const float* bk    = (const float*)d_in[8];
  const float* Wv    = (const float*)d_in[9];
  const float* bv    = (const float*)d_in[10];
  const float* Wo    = (const float*)d_in[11];
  const float* bo    = (const float*)d_in[12];
  const float* W1    = (const float*)d_in[13];
  const float* b1    = (const float*)d_in[14];
  const float* W2    = (const float*)d_in[15];
  const float* b2    = (const float*)d_in[16];

  char* ws = (char*)d_ws;
  size_t off = 0;
  auto alloc = [&](size_t bytes) -> void* {
    void* p = ws + off;
    off = (off + bytes + 255) & ~(size_t)255;
    return p;
  };

  __bf16* WqkvT = (__bf16*)alloc((size_t)CQ3 * CD * 2);
  float*  bqkv  = (float*)alloc((size_t)CQ3 * 4);
  __bf16* WoT   = (__bf16*)alloc((size_t)CD * CD * 2);
  __bf16* W1T   = (__bf16*)alloc((size_t)CFF * CD * 2);
  __bf16* W2T   = (__bf16*)alloc((size_t)CD * CFF * 2);
  __bf16* h_b   = (__bf16*)alloc((size_t)CM * CD * 2);    // reused for LN2
  __bf16* qkv_b = (__bf16*)alloc((size_t)CM * CQ3 * 2);   // q|k used
  __bf16* vt_b  = (__bf16*)alloc((size_t)CB * CH * CHD * CS * 2);
  __bf16* o_b   = (__bf16*)alloc((size_t)CM * CD * 2);
  float*  x1    = (float*)alloc((size_t)CM * CD * 4);
  __bf16* ff_b  = (__bf16*)alloc((size_t)CM * CFF * 2);

  // Weight prep: bf16 + transpose to [N,K] (q part pre-scaled by 1/sqrt(HD)).
  cvt_transpose_kernel<<<2048, 256, 0, stream>>>(Wo, WoT, CD, CD);
  cvt_transpose_kernel<<<2048, 256, 0, stream>>>(W1, W1T, CD, CFF);
  cvt_transpose_kernel<<<2048, 256, 0, stream>>>(W2, W2T, CFF, CD);
  build_wqkvT_kernel<<<2048, 256, 0, stream>>>(Wq, Wk, Wv, WqkvT);
  build_bqkv_kernel<<<12, 256, 0, stream>>>(bq, bk, bv, bqkv);

  // h = LN1(x)
  layernorm_bf16_kernel<<<CM, 256, 0, stream>>>(x, ln1_g, ln1_b, h_b);

  // qkv = h * Wqkv + bqkv ; q,k -> qkv_b, v -> vt_b (transposed per head)
  gemm_bt_kernel<3, __bf16><<<dim3(CQ3 / 128, CM / 128), 256, 0, stream>>>(
      h_b, WqkvT, bqkv, nullptr, qkv_b, vt_b, CM, CQ3, CD);

  // attention -> o_b [M, 1024]
  attn_kernel<<<dim3(CS / 128, CB * CH), 256, 0, stream>>>(qkv_b, vt_b, o_b);

  // x1 = x + o * Wo + bo
  gemm_bt_kernel<2, float><<<dim3(CD / 128, CM / 128), 256, 0, stream>>>(
      o_b, WoT, bo, x, x1, nullptr, CM, CD, CD);

  // h2 = LN2(x1)
  layernorm_bf16_kernel<<<CM, 256, 0, stream>>>(x1, ln2_g, ln2_b, h_b);

  // ff = gelu(h2 * W1 + b1)
  gemm_bt_kernel<1, __bf16><<<dim3(CFF / 128, CM / 128), 256, 0, stream>>>(
      h_b, W1T, b1, nullptr, ff_b, nullptr, CM, CFF, CD);

  // out = x1 + ff * W2 + b2
  gemm_bt_kernel<2, float><<<dim3(CD / 128, CM / 128), 256, 0, stream>>>(
      ff_b, W2T, b2, x1, (float*)d_out, nullptr, CM, CD, CFF);
}
